// CompILE_67843303408189
// MI455X (gfx1250) — compile-verified, matrix operand order checked by reading, not run on hardware
//
#include <hip/hip_runtime.h>
#include <hip/hip_bf16.h>
#include <math.h>

#define T_SEQ 2048
#define H_DIM 1024
#define V_DIM 2048
#define L_DIM 128
#define S_SEG 4
#define G4    (4 * H_DIM)          // 4096 gate outputs
#define LSTM_BLOCKS 32             // one WGP each; 128 gate rows per block
#define LSTM_THREADS 128
#define WROWS (G4 / LSTM_BLOCKS)   // 128 rows per block
#define WROW_U32 (H_DIM / 2)       // 512 packed-bf16 dwords per row
#define LSTM_LDS_BYTES (WROWS * WROW_U32 * 4 + H_DIM * 4)   // 256KB weights + 4KB h

typedef __attribute__((ext_vector_type(16))) __bf16 bf16x16;
typedef __attribute__((ext_vector_type(8)))  float  f32x8;
typedef __attribute__((ext_vector_type(4)))  unsigned int u32x4;
typedef __attribute__((ext_vector_type(8)))  int    i32x8;
typedef __attribute__((ext_vector_type(4)))  int    i32x4;

// ---------------- helpers ----------------

__device__ __forceinline__ unsigned short f2bf(float f) {
    union { float f; unsigned u; } c; c.f = f;
    unsigned u = c.u;
    unsigned r = u + 0x7fffu + ((u >> 16) & 1u);       // round-to-nearest-even
    if ((u & 0x7f800000u) == 0x7f800000u) r = u;       // inf/nan passthrough
    return (unsigned short)(r >> 16);
}

__device__ __forceinline__ float bf_lo(unsigned u) {
    union { unsigned u; float f; } c; c.u = u << 16; return c.f;
}
__device__ __forceinline__ float bf_hi(unsigned u) {
    union { unsigned u; float f; } c; c.u = u & 0xffff0000u; return c.f;
}

__device__ __forceinline__ unsigned mix32(unsigned x) {
    x ^= x >> 17; x *= 0xed5ad4bbu;
    x ^= x >> 11; x *= 0xac4c1b51u;
    x ^= x >> 15; x *= 0x31848babu;
    x ^= x >> 14;
    return x;
}

__device__ __forceinline__ float u01(unsigned a, unsigned b, unsigned c) {
    unsigned h = mix32(a * 0x9E3779B9u ^ mix32(b + 0x7f4a7c15u) ^ (c * 0x85ebca6bu));
    return (float)(h >> 8) * (1.0f / 16777216.0f) + 1e-12f;
}

__device__ __forceinline__ float sigmoidf(float x) { return 1.0f / (1.0f + expf(-x)); }

// ---------------- data prep kernels ----------------

__global__ void gather_embed(const int* __restrict__ idx, const float* __restrict__ embw,
                             float* __restrict__ embf, unsigned short* __restrict__ embb) {
    int t = blockIdx.x;
    int row = idx[t];
    const float* src = embw + (size_t)row * H_DIM;
    float* df = embf + (size_t)t * H_DIM;
    unsigned short* db = embb + (size_t)t * H_DIM;
    for (int i = threadIdx.x; i < H_DIM; i += blockDim.x) {
        float v = src[i];
        df[i] = v;
        db[i] = f2bf(v);
    }
}

__global__ void cvt_bf16(const float* __restrict__ s, unsigned short* __restrict__ d, int n) {
    int i = blockIdx.x * blockDim.x + threadIdx.x;
    if (i < n) d[i] = f2bf(s[i]);
}

// pack two consecutive f32 into one dword of two bf16 (lo = even idx, hi = odd idx)
__global__ void pack_bf16x2(const float* __restrict__ s, unsigned* __restrict__ d, int npairs) {
    int i = blockIdx.x * blockDim.x + threadIdx.x;
    if (i < npairs)
        d[i] = (unsigned)f2bf(s[2 * i]) | ((unsigned)f2bf(s[2 * i + 1]) << 16);
}

__global__ void bias_sum(const float* __restrict__ a, const float* __restrict__ b,
                         float* __restrict__ o, int n) {
    int i = blockIdx.x * blockDim.x + threadIdx.x;
    if (i < n) o[i] = a[i] + b[i];
}

__global__ void init_mask(float* __restrict__ mask, float* __restrict__ logacc, int n) {
    int i = blockIdx.x * blockDim.x + threadIdx.x;
    if (i < n) { mask[i] = 1.0f; logacc[i] = 0.0f; }
}

__global__ void init_seg(float* __restrict__ h, float* __restrict__ c, unsigned* __restrict__ bar) {
    int i = blockIdx.x * blockDim.x + threadIdx.x;
    if (i < H_DIM) { h[i] = 0.0f; c[i] = 0.0f; }
    if (i == 0) { bar[0] = 0u; bar[1] = 0u; }
}

// ---------------- WMMA GEMM: C[M,N] = act(A[M,K] * B[N,K]^T + bias) (bf16 in, f32 out) --------
// block = 256 threads = 8 waves, each wave owns one 16x16 tile; block tile = 32(M) x 64(N).
// K is a compile-time constant so the double-buffered pipeline is branch-free:
//   load(k+32) -> wmma(k) -> load(k+64) -> wmma(k+32), epilogue peeled.

union Frag { bf16x16 v; uint4 q[2]; };

template <int K>
__global__ __launch_bounds__(256) void wmma_gemm_bf16(
        const unsigned short* __restrict__ A, const unsigned short* __restrict__ B,
        const float* __restrict__ bias, float* __restrict__ C,
        int N, int relu) {
    const int lane  = threadIdx.x & 31;
    const int wid   = threadIdx.x >> 5;
    const int half  = lane >> 4;      // 0: lanes 0-15, 1: lanes 16-31
    const int lr    = lane & 15;
    const int tileM = blockIdx.y * 32 + (wid >> 2) * 16;
    const int tileN = blockIdx.x * 64 + (wid & 3) * 16;

    const unsigned short* arow = A + (size_t)(tileM + lr) * K;
    const unsigned short* brow = B + (size_t)(tileN + lr) * K;

    // 16-bit A 16x32 fragment (ISA 05_wmma layout): lanes<16 hold K {kb..kb+7, kb+16..kb+23}
    auto loadfrag = [&](Frag& a, Frag& b, int kk) {
        int kb = kk + half * 8;
        a.q[0] = *reinterpret_cast<const uint4*>(arow + kb);
        a.q[1] = *reinterpret_cast<const uint4*>(arow + kb + 16);
        b.q[0] = *reinterpret_cast<const uint4*>(brow + kb);
        b.q[1] = *reinterpret_cast<const uint4*>(brow + kb + 16);
    };

    f32x8 acc = {0.f, 0.f, 0.f, 0.f, 0.f, 0.f, 0.f, 0.f};
    Frag a0, b0, a1, b1;
    loadfrag(a0, b0, 0);
    #pragma unroll
    for (int kk = 0; kk < K - 64; kk += 64) {
        loadfrag(a1, b1, kk + 32);
        acc = __builtin_amdgcn_wmma_f32_16x16x32_bf16(
                  false, a0.v, false, b0.v, (short)0, acc, false, false);
        loadfrag(a0, b0, kk + 64);
        acc = __builtin_amdgcn_wmma_f32_16x16x32_bf16(
                  false, a1.v, false, b1.v, (short)0, acc, false, false);
    }
    loadfrag(a1, b1, K - 32);
    acc = __builtin_amdgcn_wmma_f32_16x16x32_bf16(
              false, a0.v, false, b0.v, (short)0, acc, false, false);
    acc = __builtin_amdgcn_wmma_f32_16x16x32_bf16(
              false, a1.v, false, b1.v, (short)0, acc, false, false);

    // C/D layout: VGPR i -> M = i (lanes 0-15) or 8+i (lanes 16-31); N = lr
    int col = tileN + lr;
    float bv = bias ? bias[col] : 0.0f;
    #pragma unroll
    for (int i = 0; i < 8; ++i) {
        int row = tileM + half * 8 + i;
        float v = acc[i] + bv;
        if (relu) v = fmaxf(v, 0.0f);
        C[(size_t)row * N + col] = v;
    }
}

// ---------------- cooperative masked-LSTM over T steps ----------------

__device__ __forceinline__ void grid_barrier(unsigned* cnt, unsigned* gen, int nblk) {
    __threadfence();
    __syncthreads();
    if (threadIdx.x == 0) {
        unsigned g = *(volatile unsigned*)gen;
        unsigned v = atomicAdd(cnt, 1u);
        if (v == (unsigned)(nblk - 1)) {
            *(volatile unsigned*)cnt = 0u;
            __threadfence();
            atomicAdd(gen, 1u);
        } else {
            while (*(volatile unsigned*)gen == g) __builtin_amdgcn_s_sleep(1);
        }
    }
    __syncthreads();
    __threadfence();
}

// grid = 32 blocks x 128 threads = 4096 threads (one per gate output).
// Each block parks its 128 rows of W_hh (packed bf16, 256KB) in LDS via the
// Tensor Data Mover, so the 8192 sequential recurrence steps never touch
// global memory for weights. 320KB-per-WGP LDS is what makes this possible.
__global__ __launch_bounds__(LSTM_THREADS) void lstm_coop(
        const float* __restrict__ xg,        // [T, 4H] precomputed x-gates (+ b_ih + b_hh)
        const unsigned* __restrict__ whh_pk, // [4H, H/2] packed bf16 pairs
        const float* __restrict__ mask,      // [T]
        float* __restrict__ h, float* __restrict__ c,
        float* __restrict__ enc,             // [T, H]
        unsigned* __restrict__ bar,
        float* __restrict__ gates) {         // [4H] scratch
    extern __shared__ __align__(16) char smem[];
    unsigned* wlds = reinterpret_cast<unsigned*>(smem);              // 256KB
    float*    hs   = reinterpret_cast<float*>(smem + WROWS * WROW_U32 * 4); // 4KB

    const int tid = threadIdx.x;
    const int j = blockIdx.x * LSTM_THREADS + tid;    // gate index 0..4095
    const unsigned* gsrc = whh_pk + (size_t)blockIdx.x * WROWS * WROW_U32;

#if __has_builtin(__builtin_amdgcn_tensor_load_to_lds)
    // TDM DMA: 2D tile, 512 dwords x 128 rows, contiguous -> LDS offset 0.
    if (tid < 32) {
        unsigned long long ga = (unsigned long long)(const void*)gsrc;
        u32x4 g0 = { 1u,                                   // count=1, user mode
                     0u,                                   // lds_addr = 0 (dynamic LDS base)
                     (unsigned)(ga & 0xffffffffu),         // global_addr[31:0]
                     (unsigned)((ga >> 32) & 0x01ffffffu) | 0x80000000u }; // addr[56:32] | type=2
        i32x8 g1 = { (int)(2u << 16),          // data_size=4B; wg_mask=0; no pad/iter/barrier
                     (int)((unsigned)WROW_U32 << 16),   // tensor_dim0[15:0]=512
                     (int)((unsigned)WROWS << 16),      // tensor_dim1[15:0]=128
                     (int)((unsigned)WROW_U32 << 16),   // tile_dim0=512
                     (int)WROWS,                        // tile_dim1=128
                     (int)WROW_U32,                     // tensor_dim0_stride=512
                     0, 0 };
        i32x4 gz4 = { 0, 0, 0, 0 };
        i32x8 gz8 = { 0, 0, 0, 0, 0, 0, 0, 0 };
        __builtin_amdgcn_tensor_load_to_lds(g0, g1, gz4, gz4, gz8, 0);
        __builtin_amdgcn_s_wait_tensorcnt(0);
    }
#else
    for (int i = tid; i < WROWS * WROW_U32; i += LSTM_THREADS) wlds[i] = gsrc[i];
#endif
    __syncthreads();

    const uint4* w4 = reinterpret_cast<const uint4*>(wlds + (size_t)tid * WROW_U32);

    for (int t = 0; t < T_SEQ; ++t) {
        for (int i = tid; i < H_DIM; i += LSTM_THREADS) hs[i] = h[i];
        __syncthreads();

        float acc = xg[(size_t)t * G4 + j];
        const float4* h4 = reinterpret_cast<const float4*>(hs);
        #pragma unroll 4
        for (int k = 0; k < WROW_U32 / 4; ++k) {          // 128 x uint4 = 512 dwords
            uint4 w = w4[k];
            float4 ha = h4[2 * k], hb = h4[2 * k + 1];
            acc += bf_lo(w.x) * ha.x + bf_hi(w.x) * ha.y;
            acc += bf_lo(w.y) * ha.z + bf_hi(w.y) * ha.w;
            acc += bf_lo(w.z) * hb.x + bf_hi(w.z) * hb.y;
            acc += bf_lo(w.w) * hb.z + bf_hi(w.w) * hb.w;
        }
        gates[j] = acc;

        grid_barrier(bar, bar + 1, LSTM_BLOCKS);

        if (j < H_DIM) {
            float gi = gates[j];
            float gf = gates[H_DIM + j];
            float gg = gates[2 * H_DIM + j];
            float go = gates[3 * H_DIM + j];
            float cn = sigmoidf(gf) * c[j] + sigmoidf(gi) * tanhf(gg);
            float hn = sigmoidf(go) * tanhf(cn);
            float m = mask[t];
            hn *= m; cn *= m;
            c[j] = cn; h[j] = hn;
            enc[(size_t)t * H_DIM + j] = hn;
        }

        grid_barrier(bar, bar + 1, LSTM_BLOCKS);
    }
}

// ---------------- generic wave-per-row GEMV: out[i] = act(W[i,:]·x + bias) ----------------
// biasMode: 0 none, 1 per-row, 2 scalar (bias[0])

__global__ __launch_bounds__(256) void gemv_k(
        const float* __restrict__ W, const float* __restrict__ x,
        const float* __restrict__ bias, int biasMode, int relu,
        float* __restrict__ out, int N, int K) {
    int lane = threadIdx.x & 31;
    int row = blockIdx.x * 8 + (threadIdx.x >> 5);
    if (row >= N) return;
    const float* w = W + (size_t)row * K;
    float acc = 0.0f;
    for (int k = lane; k < K; k += 32) acc += w[k] * x[k];
    for (int off = 16; off; off >>= 1) acc += __shfl_xor(acc, off, 32);
    if (lane == 0) {
        float b = (biasMode == 1) ? bias[row] : ((biasMode == 2) ? bias[0] : 0.0f);
        float v = acc + b;
        if (relu) v = fmaxf(v, 0.0f);
        out[row] = v;
    }
}

// ---------------- gumbel softmax over T (single block, 1024 threads) ----------------

__global__ __launch_bounds__(1024) void gumbel_softmax(
        const float* __restrict__ logits, float* __restrict__ sb, unsigned seed) {
    __shared__ float s[T_SEQ];
    __shared__ float red[1024];
    int tid = threadIdx.x;
    for (int i = tid; i < T_SEQ; i += 1024) {
        float lg = (i == 0) ? -1e30f : logits[i];
        float u = u01(seed, (unsigned)i, 0x1234u);
        float gum = -logf(-logf(u + 1e-16f) + 1e-16f);
        s[i] = lg + gum;                       // TEMP_B = 1
    }
    __syncthreads();
    float m = fmaxf(s[tid], s[tid + 1024]);
    red[tid] = m;
    __syncthreads();
    for (int off = 512; off; off >>= 1) {
        if (tid < off) red[tid] = fmaxf(red[tid], red[tid + off]);
        __syncthreads();
    }
    float mx = red[0];
    __syncthreads();
    float e0 = expf(s[tid] - mx), e1 = expf(s[tid + 1024] - mx);
    red[tid] = e0 + e1;
    __syncthreads();
    for (int off = 512; off; off >>= 1) {
        if (tid < off) red[tid] += red[tid + off];
        __syncthreads();
    }
    float inv = 1.0f / red[0];
    sb[tid] = e0 * inv;
    sb[tid + 1024] = e1 * inv;
}

__global__ void onehot_last(float* __restrict__ sb) {
    int i = blockIdx.x * blockDim.x + threadIdx.x;
    if (i < T_SEQ) sb[i] = (i == T_SEQ - 1) ? 1.0f : 0.0f;
}

// ---------------- readout: out[h] = sum_{t<T-1} enc[t,h]*sb[t+1] ----------------

__global__ void readout_k(const float* __restrict__ enc, const float* __restrict__ sb,
                          float* __restrict__ out) {
    int h = blockIdx.x * blockDim.x + threadIdx.x;
    if (h >= H_DIM) return;
    float acc = 0.0f;
    for (int t = 0; t < T_SEQ - 1; ++t) acc += enc[(size_t)t * H_DIM + h] * sb[t + 1];
    out[h] = acc;
}

// ---------------- z sampling (Box-Muller from hash) ----------------

__global__ void sample_z_k(const float* __restrict__ lz, float* __restrict__ z, unsigned seed) {
    int i = threadIdx.x;
    if (i >= L_DIM) return;
    float mu = lz[i], lv = lz[L_DIM + i];
    float u1 = u01(seed, (unsigned)i, 1u);
    float u2 = u01(seed, (unsigned)i, 2u);
    float n = sqrtf(-2.0f * logf(u1)) * cosf(6.2831853f * u2);
    z[i] = mu + n * expf(0.5f * lv);
}

// ---------------- mask update: logacc += log(cumsum(sb)+eps); mask = exp(logacc) ---------

__global__ __launch_bounds__(1024) void scan_mask(
        const float* __restrict__ sb, float* __restrict__ logacc, float* __restrict__ mask) {
    __shared__ float s[T_SEQ];
    int tid = threadIdx.x;
    s[tid] = sb[tid];
    s[tid + 1024] = sb[tid + 1024];
    __syncthreads();
    for (int off = 1; off < T_SEQ; off <<= 1) {
        float a0 = s[tid];
        float a1 = s[tid + 1024];
        float b0 = (tid >= off) ? s[tid - off] : 0.0f;
        float b1 = (tid + 1024 >= off) ? s[tid + 1024 - off] : 0.0f;
        __syncthreads();
        s[tid] = a0 + b0;
        s[tid + 1024] = a1 + b1;
        __syncthreads();
    }
    for (int i = tid; i < T_SEQ; i += 1024) {
        float la = logacc[i] + logf(s[i] + 1e-16f);
        logacc[i] = la;
        mask[i] = expf(la);
    }
}

// ---------------- broadcast pred[V] over T into output segment ----------------

__global__ void bcast_pred(const float* __restrict__ pred, float* __restrict__ out, int n4) {
    int idx = blockIdx.x * blockDim.x + threadIdx.x;
    if (idx >= n4) return;
    int e = idx * 4;
    int v = e & (V_DIM - 1);
    float4 p = *reinterpret_cast<const float4*>(pred + v);
    *reinterpret_cast<float4*>(out + e) = p;
}

// ---------------- host orchestration ----------------

extern "C" void kernel_launch(void* const* d_in, const int* in_sizes, int n_in,
                              void* d_out, int out_size, void* d_ws, size_t ws_size,
                              hipStream_t stream) {
    const int*   inputs  = (const int*)  d_in[0];
    const float* embed_w = (const float*)d_in[1];
    const float* w_ih    = (const float*)d_in[2];
    const float* w_hh    = (const float*)d_in[3];
    const float* b_ih    = (const float*)d_in[4];
    const float* b_hh    = (const float*)d_in[5];
    const float* wb1     = (const float*)d_in[6];
    const float* bb1     = (const float*)d_in[7];
    const float* wb2     = (const float*)d_in[8];
    const float* bb2     = (const float*)d_in[9];
    const float* wz1     = (const float*)d_in[10];
    const float* bz1     = (const float*)d_in[11];
    const float* wz2     = (const float*)d_in[12];
    const float* bz2     = (const float*)d_in[13];
    const float* wd1     = (const float*)d_in[14];
    const float* bd1     = (const float*)d_in[15];
    const float* wd2     = (const float*)d_in[16];
    const float* bd2     = (const float*)d_in[17];
    float* out = (float*)d_out;

    char* ws = (char*)d_ws;
    size_t off = 0;
    auto alloc = [&](size_t bytes) -> void* {
        off = (off + 255) & ~(size_t)255;
        void* p = ws + off;
        off += bytes;
        return p;
    };

    float*          emb_f   = (float*)         alloc((size_t)T_SEQ * H_DIM * 4);
    unsigned short* emb_b   = (unsigned short*)alloc((size_t)T_SEQ * H_DIM * 2);
    unsigned short* wih_b   = (unsigned short*)alloc((size_t)G4 * H_DIM * 2);
    unsigned short* wb1_b   = (unsigned short*)alloc((size_t)H_DIM * H_DIM * 2);
    unsigned*       whh_pk  = (unsigned*)      alloc((size_t)G4 * WROW_U32 * 4);
    float*          xg      = (float*)         alloc((size_t)T_SEQ * G4 * 4);
    float*          enc     = (float*)         alloc((size_t)T_SEQ * H_DIM * 4);
    unsigned short* enc_b   = (unsigned short*)alloc((size_t)T_SEQ * H_DIM * 2);
    float*          hid     = (float*)         alloc((size_t)T_SEQ * H_DIM * 4);
    float*          logits  = (float*)         alloc(T_SEQ * 4);
    float*          sb      = (float*)         alloc(T_SEQ * 4);
    float*          logacc  = (float*)         alloc(T_SEQ * 4);
    float*          maskv   = (float*)         alloc(T_SEQ * 4);
    float*          bsum    = (float*)         alloc(G4 * 4);
    float*          gates   = (float*)         alloc(G4 * 4);
    float*          hvec    = (float*)         alloc(H_DIM * 4);
    float*          cvec    = (float*)         alloc(H_DIM * 4);
    float*          rdout   = (float*)         alloc(H_DIM * 4);
    float*          hz      = (float*)         alloc(H_DIM * 4);
    float*          lz      = (float*)         alloc(2 * L_DIM * 4);
    float*          zvec    = (float*)         alloc(L_DIM * 4);
    float*          hd      = (float*)         alloc(H_DIM * 4);
    float*          pred    = (float*)         alloc(V_DIM * 4);
    unsigned*       bar     = (unsigned*)      alloc(256);

    // permit 260KB dynamic LDS for the persistent LSTM kernel (CDNA5: 320KB/WG)
    (void)hipFuncSetAttribute((const void*)lstm_coop,
                              hipFuncAttributeMaxDynamicSharedMemorySize,
                              LSTM_LDS_BYTES);

    // ---- segment-invariant precompute ----
    gather_embed<<<T_SEQ, 256, 0, stream>>>(inputs, embed_w, emb_f, emb_b);
    cvt_bf16<<<(G4 * H_DIM + 255) / 256, 256, 0, stream>>>(w_ih, wih_b, G4 * H_DIM);
    cvt_bf16<<<(H_DIM * H_DIM + 255) / 256, 256, 0, stream>>>(wb1, wb1_b, H_DIM * H_DIM);
    pack_bf16x2<<<(G4 * WROW_U32 + 255) / 256, 256, 0, stream>>>(w_hh, whh_pk, G4 * WROW_U32);
    bias_sum<<<(G4 + 255) / 256, 256, 0, stream>>>(b_ih, b_hh, bsum, G4);

    // x-side gates for all T (same for every segment): [T,4H] = emb @ w_ih^T + (b_ih+b_hh)
    {
        dim3 g(G4 / 64, T_SEQ / 32);
        wmma_gemm_bf16<H_DIM><<<g, 256, 0, stream>>>(emb_b, wih_b, bsum, xg, G4, 0);
    }
    init_mask<<<(T_SEQ + 255) / 256, 256, 0, stream>>>(maskv, logacc, T_SEQ);

    for (int seg = 0; seg < S_SEG; ++seg) {
        // masked LSTM over T (sequential recurrence, grid-synced cooperative kernel,
        // W_hh resident in LDS via TDM)
        init_seg<<<4, 256, 0, stream>>>(hvec, cvec, bar);
        lstm_coop<<<LSTM_BLOCKS, LSTM_THREADS, LSTM_LDS_BYTES, stream>>>(
            xg, whh_pk, maskv, hvec, cvec, enc, bar, gates);
        cvt_bf16<<<((int)((size_t)T_SEQ * H_DIM) + 255) / 256, 256, 0, stream>>>(
            enc, enc_b, T_SEQ * H_DIM);

        // boundaries b
        if (seg < S_SEG - 1) {
            dim3 g(H_DIM / 64, T_SEQ / 32);
            wmma_gemm_bf16<H_DIM><<<g, 256, 0, stream>>>(enc_b, wb1_b, bb1, hid, H_DIM, 1);
            gemv_k<<<T_SEQ / 8, 256, 0, stream>>>(hid, wb2, bb2, 2, 0, logits,
                                                  T_SEQ, H_DIM);
            gumbel_softmax<<<1, 1024, 0, stream>>>(logits, sb, 0xB0D00000u + seg);
        } else {
            onehot_last<<<(T_SEQ + 255) / 256, 256, 0, stream>>>(sb);
        }

        // latents z
        readout_k<<<H_DIM / 256, 256, 0, stream>>>(enc, sb, rdout);
        gemv_k<<<H_DIM / 8, 256, 0, stream>>>(wz1, rdout, bz1, 1, 1, hz, H_DIM, H_DIM);
        gemv_k<<<(2 * L_DIM + 7) / 8, 256, 0, stream>>>(wz2, hz, bz2, 1, 0, lz,
                                                        2 * L_DIM, H_DIM);
        sample_z_k<<<1, 128, 0, stream>>>(lz, zvec, 0x5EED0000u + seg);

        // next mask
        if (seg < S_SEG - 1) scan_mask<<<1, 1024, 0, stream>>>(sb, logacc, maskv);

        // decoder + broadcast over T
        gemv_k<<<H_DIM / 8, 256, 0, stream>>>(wd1, zvec, bd1, 1, 1, hd, H_DIM, L_DIM);
        gemv_k<<<V_DIM / 8, 256, 0, stream>>>(wd2, hd, bd2, 1, 0, pred, V_DIM, H_DIM);
        int n4 = T_SEQ * V_DIM / 4;
        bcast_pred<<<(n4 + 255) / 256, 256, 0, stream>>>(
            pred, out + (size_t)seg * T_SEQ * V_DIM, n4);
    }
}
